// resnet_cam_88450556494049
// MI455X (gfx1250) — matrix operand order repaired and checked by reading, not compiled, not run on hardware
//
#include <hip/hip_runtime.h>

typedef unsigned short ushort_t;
typedef __attribute__((ext_vector_type(16))) __bf16 bf16x16;
typedef __attribute__((ext_vector_type(8)))  float  f32x8;
typedef __attribute__((ext_vector_type(4)))  unsigned int u32x4;

#define BN_EPS 1e-5f
#define TILE_M 64
#define TILE_N 128
#define TILE_K 32
#define KPAD   40   // padded row stride (elements) for LDS tiles

enum { CF_RELU = 1, CF_OUTF32 = 2 };

__device__ __forceinline__ ushort_t f2bf(float f) {
  unsigned int u = __builtin_bit_cast(unsigned int, f);
  u += 0x7fffu + ((u >> 16) & 1u);           // round-to-nearest-even
  return (ushort_t)(u >> 16);
}
__device__ __forceinline__ float bf2f(ushort_t h) {
  unsigned int u = ((unsigned int)h) << 16;
  return __builtin_bit_cast(float, u);
}

// ---------------- elementwise fp32 -> bf16 ----------------
__global__ void f32_to_bf16_kernel(const float* __restrict__ in,
                                   ushort_t* __restrict__ out, int n) {
  int i = blockIdx.x * blockDim.x + threadIdx.x;
  if (i < n) out[i] = f2bf(in[i]);
}

// ------------- fold BN into bf16 weights + bias -------------
__global__ void bnfold_kernel(const float* __restrict__ w,
                              const float* __restrict__ g,
                              const float* __restrict__ bb,
                              const float* __restrict__ m,
                              const float* __restrict__ v,
                              const float* __restrict__ bsrc,
                              ushort_t* __restrict__ wbf,
                              float* __restrict__ bias,
                              int O, int K) {
  int i = blockIdx.x * blockDim.x + threadIdx.x;
  int total = O * K;
  if (i < total) {
    int o = i / K;
    float scale = g ? (g[o] * rsqrtf(v[o] + BN_EPS)) : 1.0f;
    wbf[i] = f2bf(w[i] * scale);
  }
  if (i < O) {
    float bsv;
    if (g) {
      float sc = g[i] * rsqrtf(v[i] + BN_EPS);
      bsv = bb[i] - m[i] * sc;
    } else {
      bsv = bsrc ? bsrc[i] : 0.0f;
    }
    bias[i] = bsv;
  }
}

// ------------- implicit-GEMM conv, bf16 WMMA, fused epilogue -------------
// D[OC, B*HO*WO] = W[OC, IC*KH*KW] x im2col(X)   (+bias, +residual, relu)
// Block: 256 threads = 8 waves; tile 64(M) x 128(N); wave tile 16(M) x 64(N).
// Double-buffered LDS, one barrier per K-step, async A-tile DMA to LDS.
__global__ void __launch_bounds__(256)
wmma_conv_kernel(const ushort_t* __restrict__ W,   // [OC, K] bf16, BN-folded
                 const float* __restrict__ bias,   // [OC]
                 const ushort_t* __restrict__ X,   // [B, IC, HI, WI] bf16
                 void* __restrict__ OUT,           // [B, OC, HO, WO] bf16 or f32
                 const ushort_t* __restrict__ RES, // residual or null
                 int NB, int IC, int HI, int WI,
                 int OC, int HO, int WO,
                 int KH, int KW, int stride, int pad, int flags) {
  __shared__ __align__(16) ushort_t ldsA[2][TILE_M * KPAD];
  __shared__ __align__(16) ushort_t ldsB[2][TILE_N * KPAD];

  const int K    = IC * KH * KW;
  const int HOWO = HO * WO;
  const int HIWI = HI * WI;
  const int Npix = NB * HOWO;
  const int nBase = blockIdx.x * TILE_N;
  const int mBase = blockIdx.y * TILE_M;
  const int t = threadIdx.x;

  // ---- A loader: 4 threads per row, 8 elems each (64 rows x 32 k) ----
  const int rowA = t >> 2;            // 0..63
  const int kqA  = (t & 3) * 8;       // 0,8,16,24
  const int  am     = mBase + rowA;
  const bool aValid = (am < OC);
  const int  wrow   = am * K;

  // ---- B loader: 2 threads per row, 16 elems each (128 rows x 32 k) ----
  const int rowB = t >> 1;            // 0..127
  const int kqB  = (t & 1) * 16;      // 0,16
  const int  bn     = nBase + rowB;
  const bool bValid = (bn < Npix);
  int ib = 0, ho_ = 0, wo_ = 0;
  if (bValid) {
    ib = bn / HOWO;
    int r = bn - ib * HOWO;
    ho_ = r / WO;
    wo_ = r - ho_ * WO;
  }
  const int hbase = ho_ * stride - pad;
  const int wbase = wo_ * stride - pad;
  const int inB   = ib * IC * HIWI;
  const int pixoff = ho_ * WI + wo_;            // 1x1 path
  const int pbase  = hbase * WI + wbase;        // KxK interior path
  const bool is1x1 = (KH == 1 && KW == 1 && pad == 0 && stride == 1);
  const bool interior = (hbase >= 0) && (hbase + KH <= HI) &&
                        (wbase >= 0) && (wbase + KW <= WI);
  const int  KHW   = KH * KW;
  const bool wfast = ((K & 7) == 0);

  // ---- wave / lane mapping ----
  const int lane = t & 31;
  const int wv   = t >> 5;            // 0..7
  const int wm   = wv & 3;            // M sub-tile (16 rows)
  const int wnq  = wv >> 2;           // N half (64 cols)
  const int lr   = lane & 15;
  const int lh   = lane >> 4;

  // stage one K-tile (k0) into LDS buffer `cb`
  auto stage = [&](int k0, int cb) {
    const bool fullK = (k0 + TILE_K) <= K;
    // A tile [64 x 32]
    if (wfast && aValid && fullK) {
      unsigned lds_off = (unsigned)(unsigned long long)
          ((__attribute__((address_space(3))) ushort_t*)&ldsA[cb][rowA * KPAD + kqA]);
      unsigned long long ga = (unsigned long long)(W + wrow + k0 + kqA);
      asm volatile("global_load_async_to_lds_b128 %0, %1, off"
                   :: "v"(lds_off), "v"(ga) : "memory");
    } else {
      union { ushort_t s[8]; u32x4 v; } tmp;
      #pragma unroll
      for (int j = 0; j < 8; ++j) {          // branchless: clamp + select
        int k = k0 + kqA + j;
        bool p = aValid && (k < K);
        int off = p ? (wrow + k) : 0;
        ushort_t val = W[off];
        tmp.s[j] = p ? val : (ushort_t)0;
      }
      *(u32x4*)(&ldsA[cb][rowA * KPAD + kqA]) = tmp.v;
    }
    // B tile transposed: ldsB[n][k] = im2col(X)[k][n]
    #pragma unroll
    for (int sub = 0; sub < 2; ++sub) {
      const int koff = kqB + sub * 8;
      union { ushort_t s[8]; u32x4 v; } tmp;
      if (bValid && fullK && (is1x1 || interior)) {
        if (is1x1) {
          const ushort_t* xp = X + inB + (k0 + koff) * HIWI + pixoff;
          #pragma unroll
          for (int j = 0; j < 8; ++j) tmp.s[j] = xp[j * HIWI];
        } else {
          #pragma unroll
          for (int j = 0; j < 8; ++j) {
            int k  = k0 + koff + j;
            int ic = k / KHW;
            int r  = k - ic * KHW;
            int kh = r / KW;
            int kw = r - kh * KW;
            tmp.s[j] = X[inB + ic * HIWI + pbase + kh * WI + kw];
          }
        }
      } else {
        #pragma unroll
        for (int j = 0; j < 8; ++j) {        // branchless: clamp + select
          int k  = k0 + koff + j;
          int kc = (k < K) ? k : 0;
          int ic = kc / KHW;
          int r  = kc - ic * KHW;
          int kh = r / KW;
          int kw = r - kh * KW;
          int hi = hbase + kh;
          int wi = wbase + kw;
          bool p = bValid && (k < K) &&
                   (hi >= 0) && (hi < HI) && (wi >= 0) && (wi < WI);
          int off = p ? (inB + (ic * HI + hi) * WI + wi) : 0;
          ushort_t val = X[off];
          tmp.s[j] = p ? val : (ushort_t)0;
        }
      }
      *(u32x4*)(&ldsB[cb][rowB * KPAD + koff]) = tmp.v;
    }
  };

  f32x8 acc[4];
  #pragma unroll
  for (int a = 0; a < 4; ++a) acc[a] = (f32x8){0,0,0,0,0,0,0,0};

  const int nk = (K + TILE_K - 1) / TILE_K;
  stage(0, 0);
  for (int i = 0; i < nk; ++i) {
    const int cb = i & 1;
    asm volatile("s_wait_asynccnt 0" ::: "memory");  // async A-tile (i) landed
    __syncthreads();                                 // tile i visible to all waves

    // ---- fragment loads for tile i (ISA 7.12.2 layouts) ----
    union FragU { struct { u32x4 lo, hi; } q; bf16x16 f; };
    FragU fa, fb[4];
    {
      // A: lane lr = row M; elems 0..7 -> K=lh*8+{0..7}, 8..15 -> K=16+lh*8+{0..7}
      const ushort_t* ar = &ldsA[cb][(wm * 16 + lr) * KPAD];
      fa.q.lo = *(const u32x4*)(ar + lh * 8);
      fa.q.hi = *(const u32x4*)(ar + 16 + lh * 8);
    }
    #pragma unroll
    for (int a = 0; a < 4; ++a) {
      // B: lane lr = col N; elems 0..7 -> K=lh*8+{0..7}, 8..15 -> K=16+lh*8+{0..7}
      const ushort_t* br = &ldsB[cb][(wnq * 64 + a * 16 + lr) * KPAD];
      fb[a].q.lo = *(const u32x4*)(br + lh * 8);
      fb[a].q.hi = *(const u32x4*)(br + 16 + lh * 8);
    }

    // ---- prefetch tile i+1 into the other buffer (overlaps with WMMA) ----
    if (i + 1 < nk) stage((i + 1) * TILE_K, cb ^ 1);

    // ---- 4x WMMA sharing the A fragment ----
    #pragma unroll
    for (int a = 0; a < 4; ++a) {
      acc[a] = __builtin_amdgcn_wmma_f32_16x16x32_bf16(
          false, fa.f, false, fb[a].f, (short)0, acc[a], false, false);
    }
  }

  // ---- epilogue: bias + optional residual + relu, store bf16 or f32 ----
  const bool relu = (flags & CF_RELU) != 0;
  const bool of32 = (flags & CF_OUTF32) != 0;
  #pragma unroll
  for (int a = 0; a < 4; ++a) {
    int n = nBase + wnq * 64 + a * 16 + lr;
    if (n < Npix) {
      int ob = n / HOWO;
      int r  = n - ob * HOWO;
      int oh = r / WO;
      int ow = r - oh * WO;
      #pragma unroll
      for (int rr = 0; rr < 8; ++rr) {
        int oc = mBase + wm * 16 + lh * 8 + rr;    // C/D layout: M = rr + 8*lh
        if (oc < OC) {
          int addr = ((ob * OC + oc) * HO + oh) * WO + ow;
          float val = acc[a][rr] + bias[oc];
          if (RES) val += bf2f(RES[addr]);
          if (relu) val = fmaxf(val, 0.0f);
          if (of32) ((float*)OUT)[addr] = val;
          else      ((ushort_t*)OUT)[addr] = f2bf(val);
        }
      }
    }
  }
}

// ---------------- maxpool 3x3 s2 p1 (bf16) ----------------
__global__ void maxpool_kernel(const ushort_t* __restrict__ in,
                               ushort_t* __restrict__ out,
                               int BC, int HI, int WI, int HO, int WO) {
  int i = blockIdx.x * blockDim.x + threadIdx.x;
  int total = BC * HO * WO;
  if (i >= total) return;
  int ow = i % WO;
  int r  = i / WO;
  int oh = r % HO;
  int bc = r / HO;
  float mx = -3.4e38f;
  int h0 = oh * 2 - 1, w0 = ow * 2 - 1;
  #pragma unroll
  for (int dh = 0; dh < 3; ++dh) {
    int h = h0 + dh;
    if (h < 0 || h >= HI) continue;
    #pragma unroll
    for (int dw = 0; dw < 3; ++dw) {
      int w = w0 + dw;
      if (w < 0 || w >= WI) continue;
      mx = fmaxf(mx, bf2f(in[(bc * HI + h) * WI + w]));
    }
  }
  out[i] = f2bf(mx);
}

// ------------- bilinear 2x upsample, align_corners=True (f32) -------------
__global__ void up2x_kernel(const float* __restrict__ in,
                            float* __restrict__ out,
                            int BC, int HI, int WI) {
  int HO = 2 * HI, WO = 2 * WI;
  int i = blockIdx.x * blockDim.x + threadIdx.x;
  int total = BC * HO * WO;
  if (i >= total) return;
  int ow = i % WO;
  int r  = i / WO;
  int oh = r % HO;
  int bc = r / HO;
  float ph = oh * (float)(HI - 1) / (float)(HO - 1);
  float pw = ow * (float)(WI - 1) / (float)(WO - 1);
  int h0 = (int)ph; int h1 = (h0 + 1 < HI) ? h0 + 1 : HI - 1; float fh = ph - h0;
  int w0 = (int)pw; int w1 = (w0 + 1 < WI) ? w0 + 1 : WI - 1; float fw = pw - w0;
  const float* p = in + bc * HI * WI;
  float v00 = p[h0 * WI + w0], v01 = p[h0 * WI + w1];
  float v10 = p[h1 * WI + w0], v11 = p[h1 * WI + w1];
  float top = v00 + (v01 - v00) * fw;
  float bot = v10 + (v11 - v10) * fw;
  out[i] = top + (bot - top) * fh;
}

// =======================================================================
extern "C" void kernel_launch(void* const* d_in, const int* in_sizes, int n_in,
                              void* d_out, int out_size, void* d_ws, size_t ws_size,
                              hipStream_t stream) {
  (void)in_sizes; (void)n_in; (void)out_size; (void)ws_size;
  int idx = 0;
  auto nextp = [&]() { return (const float*)d_in[idx++]; };
  const float* x = nextp();

  // deterministic bump allocator over workspace
  char* wsb = (char*)d_ws;
  size_t off = 0;
  auto walloc = [&](size_t bytes) -> void* {
    void* p = wsb + off;
    off = (off + bytes + 255) & ~(size_t)255;
    return p;
  };

  const int NB = 4;
  const size_t ACT_ELEMS = (size_t)4 * 64 * 112 * 112;  // largest activation
  ushort_t* bufs[4];
  for (int i = 0; i < 4; ++i) bufs[i] = (ushort_t*)walloc(ACT_ELEMS * 2);
  ushort_t* xbf  = (ushort_t*)walloc((size_t)4 * 3 * 224 * 224 * 2);
  float*    cam0 = (float*)walloc((size_t)4 * 21 * 7 * 7 * 4);
  float*    cam1 = (float*)walloc((size_t)4 * 21 * 14 * 14 * 4);
  float*    cam2 = (float*)walloc((size_t)4 * 21 * 28 * 28 * 4);

  auto run_conv = [&](const float* w, const float* g, const float* bbn,
                      const float* m, const float* v, const float* bsrc,
                      const ushort_t* in, void* outp, const ushort_t* res,
                      int IC, int HI, int WI, int OC,
                      int KH, int KW, int stride, int pad,
                      int flags, int HO, int WO) {
    int K = IC * KH * KW;
    ushort_t* wbf  = (ushort_t*)walloc((size_t)OC * K * 2);
    float*    bias = (float*)walloc((size_t)OC * 4);
    int tot = OC * K;
    bnfold_kernel<<<(tot + 255) / 256, 256, 0, stream>>>(
        w, g, bbn, m, v, bsrc, wbf, bias, OC, K);
    int Npix = NB * HO * WO;
    dim3 grid((Npix + TILE_N - 1) / TILE_N, (OC + TILE_M - 1) / TILE_M);
    wmma_conv_kernel<<<grid, 256, 0, stream>>>(
        wbf, bias, in, outp, res,
        NB, IC, HI, WI, OC, HO, WO, KH, KW, stride, pad, flags);
  };

  // ---- stem: x -> bf16, conv7x7 s2 p3 + BN + relu, maxpool 3x3 s2 p1 ----
  {
    int n = 4 * 3 * 224 * 224;
    f32_to_bf16_kernel<<<(n + 255) / 256, 256, 0, stream>>>(x, xbf, n);
  }
  const float* stem_w = nextp();
  const float* sg = nextp(); const float* sb = nextp();
  const float* sm = nextp(); const float* sv = nextp();
  run_conv(stem_w, sg, sb, sm, sv, nullptr, xbf, bufs[0], nullptr,
           3, 224, 224, 64, 7, 7, 2, 3, CF_RELU, 112, 112);
  {
    int tot = 4 * 64 * 56 * 56;
    maxpool_kernel<<<(tot + 255) / 256, 256, 0, stream>>>(
        bufs[0], bufs[1], 4 * 64, 112, 112, 56, 56);
  }

  // ---- bottleneck layers ----
  int curi = 1;
  int H = 56, W = 56;
  int inplanes = 64;
  const int planes_arr[4] = {64, 128, 256, 512};
  const int blocks_arr[4] = {3, 4, 23, 3};
  const int stride_arr[4] = {1, 2, 2, 2};
  struct CP { const float *w, *g, *b, *m, *v; };
  auto rdc = [&]() {
    CP c; c.w = nextp(); c.g = nextp(); c.b = nextp(); c.m = nextp(); c.v = nextp();
    return c;
  };
  for (int l = 0; l < 4; ++l) {
    int planes = planes_arr[l];
    for (int bi = 0; bi < blocks_arr[l]; ++bi) {
      int s = (bi == 0) ? stride_arr[l] : 1;
      CP c1 = rdc(), c2 = rdc(), c3 = rdc();
      bool has_ds = (bi == 0) && (s != 1 || inplanes != planes * 4);
      CP cd{};
      if (has_ds) cd = rdc();
      int H2 = (H - 1) / s + 1, W2 = (W - 1) / s + 1;
      int o1 = (curi + 1) & 3, o2 = (curi + 2) & 3, o3 = (curi + 3) & 3;
      // conv1 1x1 + relu
      run_conv(c1.w, c1.g, c1.b, c1.m, c1.v, nullptr, bufs[curi], bufs[o1], nullptr,
               inplanes, H, W, planes, 1, 1, 1, 0, CF_RELU, H, W);
      // conv2 3x3 (stride s, pad 1) + relu
      run_conv(c2.w, c2.g, c2.b, c2.m, c2.v, nullptr, bufs[o1], bufs[o2], nullptr,
               planes, H, W, planes, 3, 3, s, 1, CF_RELU, H2, W2);
      // shortcut
      const ushort_t* res;
      if (has_ds) {
        run_conv(cd.w, cd.g, cd.b, cd.m, cd.v, nullptr, bufs[curi], bufs[o3], nullptr,
                 inplanes, H, W, planes * 4, 1, 1, s, 0, 0, H2, W2);
        res = bufs[o3];
      } else {
        res = bufs[curi];
      }
      // conv3 1x1 + residual add + relu (fused)
      run_conv(c3.w, c3.g, c3.b, c3.m, c3.v, nullptr, bufs[o2], bufs[o1], res,
               planes, H2, W2, planes * 4, 1, 1, 1, 0, CF_RELU, H2, W2);
      curi = o1;
      inplanes = planes * 4;
      H = H2; W = W2;
    }
  }

  // ---- CAM head: Linear(2048->21) as 1x1 conv, f32 output ----
  const float* fcw = nextp();
  const float* fcb = nextp();
  run_conv(fcw, nullptr, nullptr, nullptr, nullptr, fcb, bufs[curi], cam0, nullptr,
           2048, 7, 7, 21, 1, 1, 1, 0, CF_OUTF32, 7, 7);

  // ---- 8x bilinear upsample (three 2x stages, align_corners) ----
  {
    int t0 = 4 * 21 * 14 * 14;
    up2x_kernel<<<(t0 + 255) / 256, 256, 0, stream>>>(cam0, cam1, 84, 7, 7);
    int t1 = 4 * 21 * 28 * 28;
    up2x_kernel<<<(t1 + 255) / 256, 256, 0, stream>>>(cam1, cam2, 84, 14, 14);
    int t2 = 4 * 21 * 56 * 56;
    up2x_kernel<<<(t2 + 255) / 256, 256, 0, stream>>>(cam2, (float*)d_out, 84, 28, 28);
  }
}
